// SelfAttentionWithAdaGN_89653147337046
// MI455X (gfx1250) — matrix operand order repaired and checked by reading, not compiled, not run on hardware
//
#include <hip/hip_runtime.h>
#include <hip/hip_bf16.h>

typedef __attribute__((ext_vector_type(16))) _Float16 v16h;
typedef __attribute__((ext_vector_type(8)))  float    v8f;
typedef unsigned int u32;
typedef u32 u32x4 __attribute__((ext_vector_type(4)));
typedef u32 u32x8 __attribute__((ext_vector_type(8)));

#define C_CH  256
#define NTOK  4096
#define NH    8
#define HD    32
#define CONDD 512

// ---- workspace byte offsets (all 1KB aligned) ----
#define WS_SCALE   0u          // 256 f32
#define WS_SHIFT   1024u       // 256 f32
#define WS_WQKV16  2048u       // 768*256 f16
#define WS_WOUT16  395264u     // 256*256 f16
#define WS_NORM16  526336u     // [4096][256] f16 token-major
#define WS_QM      2623488u    // [8][4096][32] f16 token-major (pre-scaled 1/16)
#define WS_KM      4720640u    // [8][4096][32] f16 token-major
#define WS_VM      6817792u    // [8][32][4096] f16 hd-major
#define WS_OBUF    8914944u    // [4096][256] f16 token-major
// total ~10.5 MB

union AV { v16h v; float4 f4[2]; };

__device__ __forceinline__ v8f wmma16(const AV& a, const AV& b, v8f c) {
  return __builtin_amdgcn_wmma_f32_16x16x32_f16(false, a.v, false, b.v,
                                                (short)0, c, false, false);
}

// ---- Tensor Data Mover: 2D f16 tile Global -> LDS (ISA 08_async_tensor §8) ----
// Descriptor group0: count=1, lds_addr, global_addr[56:0], type=2.
// Descriptor group1: data_size=2B, tensor dims, tile dims, dim0 stride.
// Rows of tile_d0 halves land contiguously in LDS (row r at lds+r*tile_d0*2).
__device__ __forceinline__ void tdm_load_2d_f16(u32 lds_byte_addr,
                                                const void* gptr,
                                                u32 tile_d0, u32 tile_d1,
                                                u32 tensor_d0, u32 tensor_d1,
                                                u32 stride0) {
  unsigned long long ga = (unsigned long long)(uintptr_t)gptr;
  u32x4 g0;
  g0.x = 1u;                                   // count=1, user, no gather
  g0.y = lds_byte_addr;                        // [63:32] lds_addr
  g0.z = (u32)ga;                              // [95:64] global_addr lo
  g0.w = (u32)((ga >> 32) & 0x01ffffffu) | (2u << 30);  // addr[56:32] | type=2
  u32x8 g1;
  g1.s0 = 1u << 16;                            // data_size=1 (2 bytes)
  g1.s1 = (tensor_d0 & 0xffffu) << 16;         // abar=0 | tensor_dim0[15:0]
  g1.s2 = (tensor_d0 >> 16) | ((tensor_d1 & 0xffffu) << 16);
  g1.s3 = (tensor_d1 >> 16) | ((tile_d0 & 0xffffu) << 16);
  g1.s4 = tile_d1 & 0xffffu;                   // tile_dim1 | tile_dim2=0
  g1.s5 = stride0;                             // tensor_dim0_stride[31:0]
  g1.s6 = 0u;                                  // stride0 hi | dim1_stride lo
  g1.s7 = 0u;
  asm volatile("tensor_load_to_lds %0, %1" :: "s"(g0), "s"(g1) : "memory");
}

// ---------------- kernel 1: AdaGN modulation from cond -------------------
__global__ void modulation_kernel(const float* __restrict__ cond,
                                  const float* __restrict__ w_scale,
                                  const float* __restrict__ b_scale,
                                  const float* __restrict__ w_shift,
                                  const float* __restrict__ b_shift,
                                  float* __restrict__ scale,
                                  float* __restrict__ shift) {
  int c = threadIdx.x;
  float a = 0.f, b = 0.f;
  for (int j = 0; j < CONDD; ++j) {
    float cv = cond[j];
    a += cv * w_scale[c * CONDD + j];
    b += cv * w_shift[c * CONDD + j];
  }
  scale[c] = a + b_scale[c];
  shift[c] = b + b_shift[c];
}

// ---------------- kernel 2: f32 -> f16 convert ---------------------------
__global__ void cvt16_kernel(const float* __restrict__ src,
                             _Float16* __restrict__ dst, int n) {
  int i = blockIdx.x * 256 + threadIdx.x;
  if (i < n) dst[i] = (_Float16)src[i];
}

// ---------------- kernel 3: group norm + AdaGN, write token-major f16 ----
__global__ void groupnorm_kernel(const float* __restrict__ x,
                                 const float* __restrict__ scale,
                                 const float* __restrict__ shift,
                                 _Float16* __restrict__ norm16) {
  __shared__ float rs[256], rs2[256];
  int g = blockIdx.x;                 // 32 groups, 8 ch x 4096 tok each
  int tid = threadIdx.x;
  const float* xb = x + (size_t)g * 8 * NTOK;
  float s = 0.f, s2 = 0.f;
  for (int i = tid; i < 8 * NTOK; i += 256) {
    float v = xb[i];
    s += v; s2 += v * v;
  }
  rs[tid] = s; rs2[tid] = s2;
  __syncthreads();
  for (int off = 128; off > 0; off >>= 1) {
    if (tid < off) { rs[tid] += rs[tid + off]; rs2[tid] += rs2[tid + off]; }
    __syncthreads();
  }
  float mean = rs[0] * (1.f / 32768.f);
  float var  = rs2[0] * (1.f / 32768.f) - mean * mean;
  float rstd = rsqrtf(var + 1e-6f);
  for (int i = tid; i < 8 * NTOK; i += 256) {
    int cl = i >> 12;           // local channel 0..7
    int t  = i & (NTOK - 1);
    int c  = g * 8 + cl;
    float v = (xb[i] - mean) * rstd;
    norm16[(size_t)t * C_CH + c] = (_Float16)(v * (1.f + scale[c]) + shift[c]);
  }
}

// ---------------- kernel 4: QKV projection GEMM (WMMA) -------------------
__global__ void qkv_gemm_kernel(const _Float16* __restrict__ W16, // [768][256]
                                const _Float16* __restrict__ X16, // [4096][256]
                                _Float16* __restrict__ Qm,
                                _Float16* __restrict__ Km,
                                _Float16* __restrict__ Vm) {
  int lane = threadIdx.x & 31, wave = threadIdx.x >> 5;
  int wid = blockIdx.x * 4 + wave;           // 12288 tiles: 48 row x 256 col
  int to = (wid % 48) * 16;
  int tt = (wid / 48) * 16;
  int hi = (lane >> 4) * 8;
  v8f acc = {};
  for (int kc = 0; kc < C_CH; kc += 32) {
    AV a, b;
    const _Float16* wr = W16 + (size_t)(to + (lane & 15)) * C_CH + kc;
    a.f4[0] = *(const float4*)(wr + hi);
    a.f4[1] = *(const float4*)(wr + 16 + hi);
    const _Float16* xr = X16 + (size_t)(tt + (lane & 15)) * C_CH + kc + (lane >> 4) * 8;
    b.f4[0] = *(const float4*)(xr);
    b.f4[1] = *(const float4*)(xr + 16);
    acc = wmma16(a, b, acc);
  }
#pragma unroll
  for (int r = 0; r < 8; ++r) {
    int o = to + r + (lane >> 4) * 8;
    int t = tt + (lane & 15);
    float v = acc[r];
    if (o < 256) {                   // Q, pre-scale by 1/sqrt(C)=1/16
      Qm[((size_t)(o >> 5) * NTOK + t) * HD + (o & 31)] = (_Float16)(v * 0.0625f);
    } else if (o < 512) {            // K token-major
      int o2 = o - 256;
      Km[((size_t)(o2 >> 5) * NTOK + t) * HD + (o2 & 31)] = (_Float16)v;
    } else {                         // V hd-major
      int o2 = o - 512;
      Vm[((size_t)(o2 >> 5) * HD + (o2 & 31)) * NTOK + t] = (_Float16)v;
    }
  }
}

// ---------------- kernel 5: flash attention (WMMA + TDM double buffer) ---
// 4 waves/block, 16 queries per wave, 32-key tiles streamed by the Tensor
// Data Mover into double-buffered LDS; wave 0 issues DMA for tile i+1 while
// all waves compute tile i.
__global__ void flash_attn_kernel(const _Float16* __restrict__ Qm,
                                  const _Float16* __restrict__ Km,
                                  const _Float16* __restrict__ Vm,
                                  _Float16* __restrict__ Ob) {
  __shared__ _Float16 kbuf[2][32 * 32];  // [key][hd]
  __shared__ _Float16 vbuf[2][32 * 32];  // [hd][key]
  __shared__ _Float16 pbuf[4 * 16 * 32]; // per-wave P tiles [q][key]
  int lane = threadIdx.x & 31, wave = threadIdx.x >> 5;
  int head = blockIdx.y;
  int m0 = blockIdx.x * 64 + wave * 16;
  int hi = (lane >> 4) * 8;

  u32 kbuf_lds[2], vbuf_lds[2];
  kbuf_lds[0] = (u32)(uintptr_t)(&kbuf[0][0]);
  kbuf_lds[1] = (u32)(uintptr_t)(&kbuf[1][0]);
  vbuf_lds[0] = (u32)(uintptr_t)(&vbuf[0][0]);
  vbuf_lds[1] = (u32)(uintptr_t)(&vbuf[1][0]);
  const _Float16* kg = Km + (size_t)head * NTOK * HD;   // rows: token, stride 32
  const _Float16* vg = Vm + (size_t)head * HD * NTOK;   // rows: hd, stride 4096

  // A-operand: 16 query rows, resident for whole kernel
  AV aq;
  const _Float16* qrow = Qm + ((size_t)head * NTOK + m0 + (lane & 15)) * HD;
  aq.f4[0] = *(const float4*)(qrow + hi);
  aq.f4[1] = *(const float4*)(qrow + 16 + hi);

  v8f O0 = {}, O1 = {};
  float mrun[8], lrun[8];
#pragma unroll
  for (int r = 0; r < 8; ++r) { mrun[r] = -1e30f; lrun[r] = 0.f; }

  const int NTILE = NTOK / 32;   // 128
  if (wave == 0) {               // prefetch tile 0
    tdm_load_2d_f16(kbuf_lds[0], kg, 32, 32, 32, NTOK, 32);
    tdm_load_2d_f16(vbuf_lds[0], vg, 32, 32, NTOK, 32, NTOK);
  }

  for (int it = 0; it < NTILE; ++it) {
    int cur = it & 1;
    if (wave == 0) {
      if (it + 1 < NTILE) {      // async DMA of next tile into other buffer
        int nk = (it + 1) * 32;
        tdm_load_2d_f16(kbuf_lds[1 - cur], kg + (size_t)nk * HD, 32, 32, 32, NTOK, 32);
        tdm_load_2d_f16(vbuf_lds[1 - cur], vg + nk, 32, 32, NTOK, 32, NTOK);
        __builtin_amdgcn_s_wait_tensorcnt(2);   // current tile's 2 DMAs done
      } else {
        __builtin_amdgcn_s_wait_tensorcnt(0);
      }
    }
    __syncthreads();             // current tile visible to all waves

    // S = Q * K^T : two 16-key tiles
    AV bk0, bk1;
    const _Float16* k0p = kbuf[cur] + (lane & 15) * 32 + (lane >> 4) * 8;
    const _Float16* k1p = k0p + 16 * 32;
    bk0.f4[0] = *(const float4*)(k0p);  bk0.f4[1] = *(const float4*)(k0p + 16);
    bk1.f4[0] = *(const float4*)(k1p);  bk1.f4[1] = *(const float4*)(k1p + 16);
    v8f zero = {};
    v8f S0 = wmma16(aq, bk0, zero);
    v8f S1 = wmma16(aq, bk1, zero);

    // online softmax; row m = r + 8*(lane/16) lives in a 16-lane group
#pragma unroll
    for (int r = 0; r < 8; ++r) {
      float mx = fmaxf(S0[r], S1[r]);
      mx = fmaxf(mx, __shfl_xor(mx, 1, 32));
      mx = fmaxf(mx, __shfl_xor(mx, 2, 32));
      mx = fmaxf(mx, __shfl_xor(mx, 4, 32));
      mx = fmaxf(mx, __shfl_xor(mx, 8, 32));
      float mnew  = fmaxf(mrun[r], mx);
      float alpha = __expf(mrun[r] - mnew);
      float p0 = __expf(S0[r] - mnew);
      float p1 = __expf(S1[r] - mnew);
      S0[r] = p0; S1[r] = p1;
      float sm = p0 + p1;
      sm += __shfl_xor(sm, 1, 32);
      sm += __shfl_xor(sm, 2, 32);
      sm += __shfl_xor(sm, 4, 32);
      sm += __shfl_xor(sm, 8, 32);
      lrun[r] = lrun[r] * alpha + sm;
      mrun[r] = mnew;
      O0[r] *= alpha;
      O1[r] *= alpha;
    }

    // D-layout P -> LDS -> A-layout reload (wave-private buffer)
    _Float16* pb = pbuf + wave * 512;
    int pr = (lane >> 4) * 8, pc = lane & 15;
#pragma unroll
    for (int r = 0; r < 8; ++r) {
      pb[(pr + r) * 32 + pc]      = (_Float16)S0[r];
      pb[(pr + r) * 32 + pc + 16] = (_Float16)S1[r];
    }
    asm volatile("s_wait_dscnt 0" ::: "memory");
    AV ap;
    const _Float16* prow = pb + (lane & 15) * 32;
    ap.f4[0] = *(const float4*)(prow + hi);
    ap.f4[1] = *(const float4*)(prow + 16 + hi);

    // O += P * V : hd 0-15 and 16-31
    AV bv0, bv1;
    const _Float16* v0p = vbuf[cur] + (lane & 15) * 32 + (lane >> 4) * 8;
    const _Float16* v1p = v0p + 16 * 32;
    bv0.f4[0] = *(const float4*)(v0p);  bv0.f4[1] = *(const float4*)(v0p + 16);
    bv1.f4[0] = *(const float4*)(v1p);  bv1.f4[1] = *(const float4*)(v1p + 16);
    O0 = wmma16(ap, bv0, O0);
    O1 = wmma16(ap, bv1, O1);

    __syncthreads();             // all waves done reading buf[cur]
  }

  // normalize and write token-major [tok][256]
#pragma unroll
  for (int r = 0; r < 8; ++r) {
    float inv = 1.0f / lrun[r];
    int tok = m0 + r + (lane >> 4) * 8;
    Ob[(size_t)tok * C_CH + head * HD + (lane & 15)]      = (_Float16)(O0[r] * inv);
    Ob[(size_t)tok * C_CH + head * HD + 16 + (lane & 15)] = (_Float16)(O1[r] * inv);
  }
}

// ---------------- kernel 6: out projection + bias + residual -------------
__global__ void outproj_kernel(const _Float16* __restrict__ W16, // [256][256]
                               const _Float16* __restrict__ Ob,  // [4096][256]
                               const float* __restrict__ b_out,
                               const float* __restrict__ x,
                               float* __restrict__ out) {
  int lane = threadIdx.x & 31, wave = threadIdx.x >> 5;
  int wid = blockIdx.x * 4 + wave;           // 4096 tiles: 16 row x 256 col
  int to = (wid & 15) * 16;
  int tt = (wid >> 4) * 16;
  int hi = (lane >> 4) * 8;
  v8f acc = {};
  for (int kc = 0; kc < C_CH; kc += 32) {
    AV a, b;
    const _Float16* wr = W16 + (size_t)(to + (lane & 15)) * C_CH + kc;
    a.f4[0] = *(const float4*)(wr + hi);
    a.f4[1] = *(const float4*)(wr + 16 + hi);
    const _Float16* xr = Ob + (size_t)(tt + (lane & 15)) * C_CH + kc + (lane >> 4) * 8;
    b.f4[0] = *(const float4*)(xr);
    b.f4[1] = *(const float4*)(xr + 16);
    acc = wmma16(a, b, acc);
  }
#pragma unroll
  for (int r = 0; r < 8; ++r) {
    int o = to + r + (lane >> 4) * 8;
    int t = tt + (lane & 15);
    out[(size_t)o * NTOK + t] = acc[r] + b_out[o] + x[(size_t)o * NTOK + t];
  }
}

extern "C" void kernel_launch(void* const* d_in, const int* in_sizes, int n_in,
                              void* d_out, int out_size, void* d_ws, size_t ws_size,
                              hipStream_t stream) {
  (void)in_sizes; (void)n_in; (void)out_size; (void)ws_size;
  const float* x       = (const float*)d_in[0];
  const float* cond    = (const float*)d_in[1];
  const float* w_scale = (const float*)d_in[2];
  const float* b_scale = (const float*)d_in[3];
  const float* w_shift = (const float*)d_in[4];
  const float* b_shift = (const float*)d_in[5];
  const float* w_qkv   = (const float*)d_in[6];
  const float* w_out   = (const float*)d_in[7];
  const float* b_out   = (const float*)d_in[8];
  float* out = (float*)d_out;

  char* ws = (char*)d_ws;
  float*    scale  = (float*)(ws + WS_SCALE);
  float*    shift  = (float*)(ws + WS_SHIFT);
  _Float16* wqkv16 = (_Float16*)(ws + WS_WQKV16);
  _Float16* wout16 = (_Float16*)(ws + WS_WOUT16);
  _Float16* norm16 = (_Float16*)(ws + WS_NORM16);
  _Float16* Qm     = (_Float16*)(ws + WS_QM);
  _Float16* Km     = (_Float16*)(ws + WS_KM);
  _Float16* Vm     = (_Float16*)(ws + WS_VM);
  _Float16* Ob     = (_Float16*)(ws + WS_OBUF);

  modulation_kernel<<<1, 256, 0, stream>>>(cond, w_scale, b_scale, w_shift, b_shift,
                                           scale, shift);
  cvt16_kernel<<<(3 * C_CH * C_CH + 255) / 256, 256, 0, stream>>>(w_qkv, wqkv16,
                                                                  3 * C_CH * C_CH);
  cvt16_kernel<<<(C_CH * C_CH + 255) / 256, 256, 0, stream>>>(w_out, wout16,
                                                              C_CH * C_CH);
  groupnorm_kernel<<<32, 256, 0, stream>>>(x, scale, shift, norm16);
  qkv_gemm_kernel<<<(48 * 256) / 4, 128, 0, stream>>>(wqkv16, norm16, Qm, Km, Vm);
  flash_attn_kernel<<<dim3(NTOK / 64, NH), 128, 0, stream>>>(Qm, Km, Vm, Ob);
  outproj_kernel<<<(16 * 256) / 4, 128, 0, stream>>>(wout16, Ob, b_out, x, out);
}
